// DecoderBlock_30451318129092
// MI455X (gfx1250) — matrix operand order repaired and checked by reading, not compiled, not run on hardware
//
#include <hip/hip_runtime.h>
#include <hip/hip_bf16.h>

// ---------------------------------------------------------------------------
// Decoder block for MI455X (gfx1250, wave32, WMMA).
// Matmuls + attention on v_wmma_f32_16x16x32_bf16 (f32 accum); LDS tiles are
// filled with GLOBAL_LOAD_ASYNC_TO_LDS_B128 (ASYNCcnt) and double-buffered.
// ---------------------------------------------------------------------------

#define B_   4
#define L_   1024
#define M_   1024
#define D_   1024
#define HD_  64
#define NH_  16
#define HID_ 4096
#define SCALE_ 0.125f   // 64^-0.5

typedef __attribute__((ext_vector_type(16))) __bf16 v16bf;
typedef __attribute__((ext_vector_type(8)))  float  v8f;

union BFrag { v16bf v; unsigned u[8]; };
union FAcc  { v8f   v; float    f[8]; };

static __device__ __forceinline__ v8f wmma_bf16(const BFrag& a, const BFrag& b, v8f c) {
    return __builtin_amdgcn_wmma_f32_16x16x32_bf16(false, a.v, false, b.v, (short)0, c,
                                                   false, false);
}

static __device__ __forceinline__ void frag_from2(BFrag& f, uint4 t0, uint4 t1) {
    f.u[0] = t0.x; f.u[1] = t0.y; f.u[2] = t0.z; f.u[3] = t0.w;
    f.u[4] = t1.x; f.u[5] = t1.y; f.u[6] = t1.z; f.u[7] = t1.w;
}

// Async 16B global -> LDS copy (CDNA5 TDM-adjacent path, tracked by ASYNCcnt).
// LDS operand is the 32-bit LDS byte offset (flat aperture keeps it in addr[31:0]).
static __device__ __forceinline__ void async_ld_lds_b128(void* lds, const void* gaddr) {
    unsigned loff = (unsigned)(size_t)lds;
    asm volatile("global_load_async_to_lds_b128 %0, %1, off"
                 :: "v"(loff), "v"(gaddr) : "memory");
}
#define WAIT_ASYNC0 asm volatile("s_wait_asynccnt 0x0" ::: "memory")
#define WAIT_ASYNC2 asm volatile("s_wait_asynccnt 0x2" ::: "memory")

static __device__ __forceinline__ float red_max16(float x) {
    #pragma unroll
    for (int off = 1; off < 16; off <<= 1) x = fmaxf(x, __shfl_xor(x, off, 16));
    return x;
}
static __device__ __forceinline__ float red_sum16(float x) {
    #pragma unroll
    for (int off = 1; off < 16; off <<= 1) x += __shfl_xor(x, off, 16);
    return x;
}

// ---------------------------------------------------------------------------
// Elementwise: f32 -> bf16 cast
// ---------------------------------------------------------------------------
__global__ void k_cast_bf16(const float* __restrict__ src, __hip_bfloat16* __restrict__ dst) {
    unsigned i = blockIdx.x * blockDim.x + threadIdx.x;
    dst[i] = __float2bfloat16(src[i]);
}

// ---------------------------------------------------------------------------
// Elementwise: exact GELU(f32) -> bf16
// ---------------------------------------------------------------------------
__global__ void k_gelu_bf16(const float* __restrict__ src, __hip_bfloat16* __restrict__ dst) {
    unsigned i = blockIdx.x * blockDim.x + threadIdx.x;
    float x = src[i];
    float g = 0.5f * x * (1.0f + erff(x * 0.70710678118654752f));
    dst[i] = __float2bfloat16(g);
}

// ---------------------------------------------------------------------------
// Row LayerNorm (D=1024) -> bf16. One 256-thread block per row, 4 elems/thread.
// ---------------------------------------------------------------------------
__global__ void k_layernorm_bf16(const float* __restrict__ x, const float* __restrict__ g,
                                 __hip_bfloat16* __restrict__ y) {
    __shared__ float s_sum[8], s_sq[8];
    const int t = threadIdx.x;
    const unsigned row = blockIdx.x;
    const float4* xr = reinterpret_cast<const float4*>(x + (size_t)row * D_);
    float4 v = xr[t];
    float sum = v.x + v.y + v.z + v.w;
    float sq  = v.x * v.x + v.y * v.y + v.z * v.z + v.w * v.w;
    #pragma unroll
    for (int off = 16; off; off >>= 1) {
        sum += __shfl_xor(sum, off, 32);
        sq  += __shfl_xor(sq,  off, 32);
    }
    const int w = t >> 5, ln = t & 31;
    if (ln == 0) { s_sum[w] = sum; s_sq[w] = sq; }
    __syncthreads();
    float tsum = 0.f, tsq = 0.f;
    #pragma unroll
    for (int i = 0; i < 8; ++i) { tsum += s_sum[i]; tsq += s_sq[i]; }
    const float mu   = tsum * (1.0f / D_);
    const float var  = tsq * (1.0f / D_) - mu * mu;
    const float rstd = rsqrtf(var + 1e-6f);
    const float4 gv = reinterpret_cast<const float4*>(g)[t];
    __hip_bfloat16* yr = y + (size_t)row * D_;
    const int c = t * 4;
    yr[c + 0] = __float2bfloat16((v.x - mu) * rstd * gv.x);
    yr[c + 1] = __float2bfloat16((v.y - mu) * rstd * gv.y);
    yr[c + 2] = __float2bfloat16((v.z - mu) * rstd * gv.z);
    yr[c + 3] = __float2bfloat16((v.w - mu) * rstd * gv.w);
}

// ---------------------------------------------------------------------------
// WMMA GEMM: C[Mr,N] = A[Mr,K] (bf16, ld=lda) x W[N,K]^T (bf16, ld=ldw)
// Block = 256 threads = 8 waves; each wave computes a 16x128 strip.
// Block tile: 128 rows x 128 cols. grid = (N/128, Mr/128).
// W tile double-buffered in LDS via global_load_async_to_lds_b128;
// layout is K-pair-interleaved dwords: ldsB[buf][n][k/2] = {k, k+1}.
// ---------------------------------------------------------------------------
template <bool F32OUT>
__global__ void k_gemm_wmma(const __hip_bfloat16* __restrict__ A, int lda,
                            const __hip_bfloat16* __restrict__ W, int ldw,
                            int K,
                            __hip_bfloat16* __restrict__ outB,
                            float* __restrict__ outF,
                            const float* __restrict__ res, int ldo) {
    __shared__ unsigned ldsB[2][128][20];  // 20-dword rows: 16B-aligned, bank-spread
    const int lane = threadIdx.x & 31;
    const int wv   = threadIdx.x >> 5;
    const int hf   = lane >> 4;
    const int n16  = lane & 15;
    const int row0 = blockIdx.y * 128 + wv * 16;
    const int col0 = blockIdx.x * 128;

    FAcc acc[8];
    #pragma unroll
    for (int c = 0; c < 8; ++c)
        #pragma unroll
        for (int e = 0; e < 8; ++e) acc[c].f[e] = 0.f;

    const unsigned aoff = (unsigned)(row0 + n16) * (unsigned)lda;
    // staging coords: thread handles rows sn0 and sn0+64, one 16B segment each
    const unsigned sn0 = threadIdx.x >> 2, sseg = (threadIdx.x & 3) * 8;
    const unsigned sn1 = sn0 + 64;

    auto stage = [&](int k0, int buf) {
        const unsigned w0 = (unsigned)(col0 + sn0) * (unsigned)ldw + k0 + sseg;
        const unsigned w1 = (unsigned)(col0 + sn1) * (unsigned)ldw + k0 + sseg;
        async_ld_lds_b128(&ldsB[buf][sn0][sseg >> 1], W + w0);
        async_ld_lds_b128(&ldsB[buf][sn1][sseg >> 1], W + w1);
    };

    stage(0, 0);   // prologue fill of buffer 0

    for (int k0 = 0; k0 < K; k0 += 32) {
        const int cur = (k0 >> 5) & 1;
        __syncthreads();               // everyone done reading the buffer we refill
        if (k0 + 32 < K) {
            stage(k0 + 32, cur ^ 1);   // overlap next tile's fill with this compute
            WAIT_ASYNC2;               // in-order: current tile's 2 copies are done
        } else {
            WAIT_ASYNC0;
        }
        __syncthreads();               // cur buffer visible block-wide

        BFrag a;
        frag_from2(a,
                   *reinterpret_cast<const uint4*>(A + aoff + k0 + hf * 8),
                   *reinterpret_cast<const uint4*>(A + aoff + k0 + 16 + hf * 8));
        if (k0 + 32 < K) __builtin_prefetch(A + aoff + k0 + 32, 0, 3);

        #pragma unroll
        for (int c = 0; c < 8; ++c) {
            BFrag b;
            const unsigned* br = &ldsB[cur][c * 16 + n16][hf * 4];
            frag_from2(b, *reinterpret_cast<const uint4*>(br),
                          *reinterpret_cast<const uint4*>(br + 8));
            acc[c].v = wmma_bf16(a, b, acc[c].v);
        }
    }

    #pragma unroll
    for (int c = 0; c < 8; ++c) {
        #pragma unroll
        for (int e = 0; e < 8; ++e) {
            const size_t row = row0 + e + 8 * hf;
            const size_t col = col0 + c * 16 + n16;
            if (F32OUT) {
                float vo = acc[c].f[e];
                if (res) vo += res[row * ldo + col];
                outF[row * ldo + col] = vo;
            } else {
                outB[row * ldo + col] = __float2bfloat16(acc[c].f[e]);
            }
        }
    }
}

// ---------------------------------------------------------------------------
// Flash attention (HD=64). Block = 8 waves sharing (batch, head); each wave
// owns a 16-row Q tile. K tile async-staged, V tile perm-packed, per j-step.
// grid = (NH, Lq/128, B), block = 256. causal => block-uniform triangular loop.
// ---------------------------------------------------------------------------
__global__ void k_attn_wmma(const __hip_bfloat16* __restrict__ Qp, int ldq,
                            const __hip_bfloat16* __restrict__ Kp, int ldk,
                            const __hip_bfloat16* __restrict__ Vp, int ldv,
                            __hip_bfloat16* __restrict__ Op, int ldo,
                            int Lq, int Lk, int causal) {
    __shared__ __hip_bfloat16 ktile[32][64];      // row-major K rows
    __shared__ unsigned       vbuf[64][20];       // vbuf[n][j/2] = {V[j][n], V[j+1][n]}
    __shared__ __hip_bfloat16 pbuf[8][16][32];    // per-wave P tile
    const int lane = threadIdx.x & 31;
    const int wv   = threadIdx.x >> 5;
    const int hf   = lane >> 4;
    const int n16  = lane & 15;
    const int head = blockIdx.x;
    const int b    = blockIdx.z;
    const int qt   = blockIdx.y * 8 + wv;
    const unsigned qrow0 = (unsigned)b * Lq + qt * 16;
    const unsigned krow0 = (unsigned)b * Lk;
    const int hoff = head * HD_;

    // Q tile as two 16x32 A-fragments (head-dim 0..31, 32..63)
    BFrag qa0, qa1;
    {
        const unsigned qoff = (qrow0 + n16) * (unsigned)ldq + hoff + hf * 16;
        frag_from2(qa0, *reinterpret_cast<const uint4*>(Qp + qoff),
                        *reinterpret_cast<const uint4*>(Qp + qoff + 8));
        frag_from2(qa1, *reinterpret_cast<const uint4*>(Qp + qoff + 32),
                        *reinterpret_cast<const uint4*>(Qp + qoff + 40));
    }

    FAcc o[4];
    #pragma unroll
    for (int c = 0; c < 4; ++c)
        #pragma unroll
        for (int e = 0; e < 8; ++e) o[c].f[e] = 0.f;
    float mrow[8], lrow[8];
    #pragma unroll
    for (int r = 0; r < 8; ++r) { mrow[r] = -__builtin_inff(); lrow[r] = 0.f; }

    const int iabs_base = qt * 16;
    int wave_jmax  = causal ? (iabs_base + 16) : Lk;
    if (wave_jmax > Lk) wave_jmax = Lk;
    int block_jmax = causal ? (blockIdx.y * 128 + 128) : Lk;
    if (block_jmax > Lk) block_jmax = Lk;

    // staging coords
    const unsigned krow_s = threadIdx.x >> 3, kseg_s = (threadIdx.x & 7) * 8;
    const unsigned vk2_s  = threadIdx.x >> 4, vcg_s  = (threadIdx.x & 15) * 4;

    for (int j0 = 0; j0 < block_jmax; j0 += 32) {
        __syncthreads();
        // --- stage K tile via async global->LDS b128
        async_ld_lds_b128(&ktile[krow_s][kseg_s],
                          Kp + (krow0 + j0 + krow_s) * (unsigned)ldk + hoff + kseg_s);
        // --- stage V tile, pair-interleaved along j via v_perm
        {
            const unsigned voff = (krow0 + j0 + 2 * vk2_s) * (unsigned)ldv + hoff + vcg_s;
            uint2 lo2 = *reinterpret_cast<const uint2*>(Vp + voff);
            uint2 hi2 = *reinterpret_cast<const uint2*>(Vp + voff + ldv);
            vbuf[vcg_s + 0][vk2_s] = __builtin_amdgcn_perm(hi2.x, lo2.x, 0x05040100u);
            vbuf[vcg_s + 1][vk2_s] = __builtin_amdgcn_perm(hi2.x, lo2.x, 0x07060302u);
            vbuf[vcg_s + 2][vk2_s] = __builtin_amdgcn_perm(hi2.y, lo2.y, 0x05040100u);
            vbuf[vcg_s + 3][vk2_s] = __builtin_amdgcn_perm(hi2.y, lo2.y, 0x07060302u);
        }
        WAIT_ASYNC0;
        __syncthreads();

        if (j0 < wave_jmax) {
            // --- S[16x32] = Q . K^T (two 16x16 chunks; K-dim 64 = 2 wmma each)
            FAcc s[2];
            #pragma unroll
            for (int jc = 0; jc < 2; ++jc) {
                #pragma unroll
                for (int e = 0; e < 8; ++e) s[jc].f[e] = 0.f;
                const __hip_bfloat16* kr = &ktile[jc * 16 + n16][0];
                BFrag kb;
                frag_from2(kb, *reinterpret_cast<const uint4*>(kr + hf * 16),
                               *reinterpret_cast<const uint4*>(kr + hf * 16 + 8));
                s[jc].v = wmma_bf16(qa0, kb, s[jc].v);
                frag_from2(kb, *reinterpret_cast<const uint4*>(kr + 32 + hf * 16),
                               *reinterpret_cast<const uint4*>(kr + 32 + hf * 16 + 8));
                s[jc].v = wmma_bf16(qa1, kb, s[jc].v);
            }

            // --- online softmax, stage P to LDS, rescale O
            #pragma unroll
            for (int r = 0; r < 8; ++r) {
                const int iabs = iabs_base + r + 8 * hf;
                float v0 = s[0].f[r] * SCALE_;
                float v1 = s[1].f[r] * SCALE_;
                if (causal && (j0 + n16)      > iabs) v0 = -__builtin_inff();
                if (causal && (j0 + 16 + n16) > iabs) v1 = -__builtin_inff();
                const float mx   = red_max16(fmaxf(v0, v1));
                const float mnew = fmaxf(mrow[r], mx);
                float sc, p0, p1;
                if (mnew == -__builtin_inff()) {
                    sc = 1.f; p0 = 0.f; p1 = 0.f;
                } else {
                    sc = (mrow[r] == -__builtin_inff()) ? 0.f : expf(mrow[r] - mnew);
                    p0 = (v0 == -__builtin_inff()) ? 0.f : expf(v0 - mnew);
                    p1 = (v1 == -__builtin_inff()) ? 0.f : expf(v1 - mnew);
                }
                const float rs = red_sum16(p0 + p1);
                lrow[r] = lrow[r] * sc + rs;
                mrow[r] = mnew;
                pbuf[wv][r + 8 * hf][n16]      = __float2bfloat16(p0);
                pbuf[wv][r + 8 * hf][16 + n16] = __float2bfloat16(p1);
                #pragma unroll
                for (int c = 0; c < 4; ++c) o[c].f[r] *= sc;
            }
            __builtin_amdgcn_wave_barrier();

            // --- reload P as 16x32 A-fragment (two b128 LDS reads)
            BFrag pa;
            {
                const __hip_bfloat16* pr = &pbuf[wv][n16][hf * 8];
                frag_from2(pa, *reinterpret_cast<const uint4*>(pr),
                               *reinterpret_cast<const uint4*>(pr + 16));
            }
            __builtin_amdgcn_wave_barrier();

            // --- O += P . V (K-dim 32)
            #pragma unroll
            for (int c = 0; c < 4; ++c) {
                BFrag vb;
                const unsigned* vr = &vbuf[c * 16 + n16][hf * 4];
                frag_from2(vb, *reinterpret_cast<const uint4*>(vr),
                               *reinterpret_cast<const uint4*>(vr + 8));
                o[c].v = wmma_bf16(pa, vb, o[c].v);
            }
        }
    }

    // --- normalize + store
    #pragma unroll
    for (int r = 0; r < 8; ++r) {
        const float inv = 1.0f / lrow[r];
        const size_t row = qrow0 + r + 8 * hf;
        #pragma unroll
        for (int c = 0; c < 4; ++c)
            Op[row * ldo + hoff + c * 16 + n16] = __float2bfloat16(o[c].f[r] * inv);
    }
}

// ---------------------------------------------------------------------------
// Host-side orchestration
// ---------------------------------------------------------------------------
extern "C" void kernel_launch(void* const* d_in, const int* in_sizes, int n_in,
                              void* d_out, int out_size, void* d_ws, size_t ws_size,
                              hipStream_t stream) {
    (void)in_sizes; (void)n_in; (void)out_size; (void)ws_size;
    const float* x    = (const float*)d_in[0];
    const float* ctx  = (const float*)d_in[1];
    // d_in[2] = sa_mask (exact causal tril; handled analytically)
    const float* Wqkv_f = (const float*)d_in[3];
    const float* Wso_f  = (const float*)d_in[4];
    const float* Wq_f   = (const float*)d_in[5];
    const float* Wkv_f  = (const float*)d_in[6];
    const float* Wco_f  = (const float*)d_in[7];
    const float* Wf1_f  = (const float*)d_in[8];
    const float* Wf2_f  = (const float*)d_in[9];
    const float* g1  = (const float*)d_in[10];
    const float* gq  = (const float*)d_in[11];
    const float* gc  = (const float*)d_in[12];
    const float* g2  = (const float*)d_in[13];

    const size_t ROWS = (size_t)B_ * L_;     // 4096 (also B*M)
    char* ws = (char*)d_ws;
    size_t off = 0;
    auto alloc = [&](size_t bytes) -> char* {
        char* p = ws + off;
        off += (bytes + 255) & ~(size_t)255;
        return p;
    };

    // bf16 weight copies
    __hip_bfloat16* Wqkv = (__hip_bfloat16*)alloc((size_t)3 * D_ * D_ * 2);
    __hip_bfloat16* Wso  = (__hip_bfloat16*)alloc((size_t)D_ * D_ * 2);
    __hip_bfloat16* Wq   = (__hip_bfloat16*)alloc((size_t)D_ * D_ * 2);
    __hip_bfloat16* Wkv  = (__hip_bfloat16*)alloc((size_t)2 * D_ * D_ * 2);
    __hip_bfloat16* Wco  = (__hip_bfloat16*)alloc((size_t)D_ * D_ * 2);
    __hip_bfloat16* Wf1  = (__hip_bfloat16*)alloc((size_t)HID_ * D_ * 2);
    __hip_bfloat16* Wf2  = (__hip_bfloat16*)alloc((size_t)D_ * HID_ * 2);
    // f32 residual streams
    float* x_a = (float*)alloc(ROWS * D_ * 4);
    float* x_b = (float*)alloc(ROWS * D_ * 4);
    // scratch regions
    char* R1 = alloc((size_t)64 << 20);   // fc1 f32 / qkv bf16 / kv bf16
    char* R2 = alloc((size_t)32 << 20);   // bf16 activations (4 x 8MiB) / gelu
    char* R2a = R2;
    char* R2b = R2 + ((size_t)8 << 20);
    char* R2c = R2 + ((size_t)16 << 20);
    char* R2d = R2 + ((size_t)24 << 20);

    // ---- weight casts
    auto cast = [&](const float* s, __hip_bfloat16* d, size_t n) {
        k_cast_bf16<<<dim3((unsigned)(n / 256)), dim3(256), 0, stream>>>(s, d);
    };
    cast(Wqkv_f, Wqkv, (size_t)3 * D_ * D_);
    cast(Wso_f,  Wso,  (size_t)D_ * D_);
    cast(Wq_f,   Wq,   (size_t)D_ * D_);
    cast(Wkv_f,  Wkv,  (size_t)2 * D_ * D_);
    cast(Wco_f,  Wco,  (size_t)D_ * D_);
    cast(Wf1_f,  Wf1,  (size_t)HID_ * D_);
    cast(Wf2_f,  Wf2,  (size_t)D_ * HID_);

    const dim3 blk(256);
    auto gemm_bf16 = [&](const __hip_bfloat16* A, int lda, const __hip_bfloat16* W, int K,
                         __hip_bfloat16* out, int N) {
        k_gemm_wmma<false><<<dim3(N / 128, (unsigned)(ROWS / 128)), blk, 0, stream>>>(
            A, lda, W, K, K, out, nullptr, nullptr, N);
    };
    auto gemm_f32 = [&](const __hip_bfloat16* A, int lda, const __hip_bfloat16* W, int K,
                        float* out, const float* res, int N) {
        k_gemm_wmma<true><<<dim3(N / 128, (unsigned)(ROWS / 128)), blk, 0, stream>>>(
            A, lda, W, K, K, nullptr, out, res, N);
    };

    // ---- self-attention
    __hip_bfloat16* h = (__hip_bfloat16*)R2a;
    k_layernorm_bf16<<<dim3((unsigned)ROWS), blk, 0, stream>>>(x, g1, h);
    __hip_bfloat16* qkv = (__hip_bfloat16*)R1;                       // [4096, 3072]
    gemm_bf16(h, D_, Wqkv, D_, qkv, 3 * D_);
    __hip_bfloat16* sa = (__hip_bfloat16*)R2b;                       // [4096, 1024]
    k_attn_wmma<<<dim3(NH_, L_ / 128, B_), blk, 0, stream>>>(
        qkv, 3 * D_, qkv + D_, 3 * D_, qkv + 2 * D_, 3 * D_, sa, D_, L_, L_, 1);
    gemm_f32(sa, D_, Wso, D_, x_a, x, D_);                           // x_a = x + sa Wso^T

    // ---- cross-attention
    __hip_bfloat16* hq = (__hip_bfloat16*)R2a;
    __hip_bfloat16* hc = (__hip_bfloat16*)R2b;
    k_layernorm_bf16<<<dim3((unsigned)ROWS), blk, 0, stream>>>(x_a, gq, hq);
    k_layernorm_bf16<<<dim3((unsigned)((size_t)B_ * M_)), blk, 0, stream>>>(ctx, gc, hc);
    __hip_bfloat16* q2 = (__hip_bfloat16*)R2c;                       // [4096, 1024]
    gemm_bf16(hq, D_, Wq, D_, q2, D_);
    __hip_bfloat16* kv = (__hip_bfloat16*)R1;                        // [4096, 2048]
    gemm_bf16(hc, D_, Wkv, D_, kv, 2 * D_);
    __hip_bfloat16* ca = (__hip_bfloat16*)R2d;
    k_attn_wmma<<<dim3(NH_, L_ / 128, B_), blk, 0, stream>>>(
        q2, D_, kv, 2 * D_, kv + D_, 2 * D_, ca, D_, L_, M_, 0);
    gemm_f32(ca, D_, Wco, D_, x_b, x_a, D_);                         // x_b = x_a + ca Wco^T

    // ---- MLP
    __hip_bfloat16* h2 = (__hip_bfloat16*)R2a;
    k_layernorm_bf16<<<dim3((unsigned)ROWS), blk, 0, stream>>>(x_b, g2, h2);
    float* fc1 = (float*)R1;                                         // [4096, 4096] f32
    gemm_f32(h2, D_, Wf1, D_, fc1, nullptr, HID_);
    __hip_bfloat16* gact = (__hip_bfloat16*)R2;                      // [4096, 4096] bf16
    k_gelu_bf16<<<dim3((unsigned)(ROWS * HID_ / 256)), blk, 0, stream>>>(fc1, gact);
    gemm_f32(gact, HID_, Wf2, HID_, (float*)d_out, x_b, D_);         // out = x_b + g Wf2^T
}
